// LlamaStyleEngramMoETransformerLM_15994458211255
// MI455X (gfx1250) — matrix operand order repaired and checked by reading, not compiled, not run on hardware
//
#include <hip/hip_runtime.h>

// ---------------------------------------------------------------------------
// Types
// ---------------------------------------------------------------------------
typedef __bf16 bf16;
typedef __attribute__((ext_vector_type(8)))  __bf16 bf16x8;
typedef __attribute__((ext_vector_type(16))) __bf16 bf16x16;
typedef __attribute__((ext_vector_type(8)))  float  f32x8;

// Model dims (compile-time constants from the reference)
#define CV 32000
#define CD 1024
#define CH 16
#define CHD 64
#define CF 2048
#define CE 8
#define CL 2
#define CB 2
#define CS 1024
#define CT (CB * CS)

// ---------------------------------------------------------------------------
// CDNA5 async Global->LDS copy (ASYNCcnt-tracked), 16B per lane.
// vdst = LDS byte offset, vaddr = 64-bit global address, GV mode (saddr=off).
// ---------------------------------------------------------------------------
__device__ __forceinline__ void async_copy16(unsigned int lds_off, const void* gaddr) {
  asm volatile("global_load_async_to_lds_b128 %0, %1, off"
               :: "v"(lds_off), "v"(gaddr) : "memory");
}
__device__ __forceinline__ void wait_async0() {
  asm volatile("s_wait_asynccnt 0x0" ::: "memory");
}
__device__ __forceinline__ unsigned int lds_off32(const void* p) {
  return (unsigned int)(unsigned long long)p; // low 32 bits of generic = LDS offset
}

// ---------------------------------------------------------------------------
// Generic batched NT GEMM:  C[b] = alpha * A[b] (M,K) * B[b]^T  (B given as
// BT row-major [N,K]).  Optional per-row scale and accumulate epilogue.
// bf16 inputs, fp32 accumulation via v_wmma_f32_16x16x32_bf16.
// Macro tile 256x64x32, 8 waves as 4(M) x 2(N), wave tile 64x32 (4x2 frags).
// Requires: M%256==0, N%64==0, K%32==0, lda/ldb multiples of 8.
// ---------------------------------------------------------------------------
__global__ __launch_bounds__(256) void k_gemm_nt(
    const bf16* __restrict__ A, long long sAb, int lda,
    const bf16* __restrict__ Bt, long long sBb, int ldb,
    float* __restrict__ C, long long sCb, int ldc,
    int M, int N, int Kd, float alpha,
    const float* __restrict__ rowScale, int rsStride, int accum)
{
  constexpr int BM = 256, BN = 64, BK = 32, LDH = 40; // 8-elem pad: conflict-free
  __shared__ bf16 sA[2][BM * LDH];
  __shared__ bf16 sB[2][BN * LDH];

  const int tid  = threadIdx.x;
  const int lane = tid & 31;
  const int wv   = tid >> 5;     // 8 waves
  const int wm   = wv & 3;       // wave M tile (0..3) -> 64 rows each
  const int wn   = wv >> 2;      // wave N tile (0..1) -> 32 cols each

  const long long bz = blockIdx.z;
  const bf16* Ab = A  + bz * sAb;
  const bf16* Bb = Bt + bz * sBb;
  float*      Cb = C  + bz * sCb;

  const int m0 = blockIdx.y * BM;
  const int n0 = blockIdx.x * BN;

  // Global->LDS copy mapping: each thread moves 16B chunks (4 for A, 1 for B).
  const int arow = tid >> 2;            // 0..63
  const int acol = (tid & 3) << 3;      // 0,8,16,24

  const bf16* gA = Ab + (size_t)(m0 + arow) * lda + acol;
  const bf16* gB = Bb + (size_t)(n0 + arow) * ldb + acol;

  // Fragment lane mapping (ISA 7.12.2)
  const int fm = lane & 15;
  const int kg = lane >> 4;

  f32x8 acc[4][2] = {};

  auto issue_tile = [&](int b, int ko) {
#pragma unroll
    for (int i = 0; i < 4; ++i)
      async_copy16(lds_off32(&sA[b][(arow + 64 * i) * LDH + acol]),
                   gA + (size_t)(64 * i) * lda + ko);
    async_copy16(lds_off32(&sB[b][arow * LDH + acol]), gB + ko);
  };

  // Prologue: stage k-tile 0 via async DMA, fence, publish.
  issue_tile(0, 0);
  wait_async0();
  __syncthreads();

  const int nk = Kd / BK;
  int buf = 0;
  for (int kt = 0; kt < nk; ++kt) {
    if (kt + 1 < nk) issue_tile(buf ^ 1, (kt + 1) * BK); // overlap with compute

    bf16x16 af[4], bfr[2];
#pragma unroll
    for (int i = 0; i < 4; ++i) {
      // A 16x32 frag: lanes 0-15 rows M, elems 0..7 = K kg*8..kg*8+7,
      // elems 8..15 = K 16+kg*8 .. 16+kg*8+7
      const bf16* p = &sA[buf][(wm * 64 + i * 16 + fm) * LDH + (kg << 3)];
      bf16x8 lo = *(const bf16x8*)p;
      bf16x8 hi = *(const bf16x8*)(p + 16);
      af[i] = __builtin_shufflevector(lo, hi, 0,1,2,3,4,5,6,7,8,9,10,11,12,13,14,15);
    }
#pragma unroll
    for (int j = 0; j < 2; ++j) {
      // B 32x16 frag: lane = column; elems 0..7 = K kg*8..kg*8+7, elems 8..15 = K 16+kg*8..16+kg*8+7
      const bf16* p = &sB[buf][(wn * 32 + j * 16 + fm) * LDH + (kg << 3)];
      bf16x8 lo = *(const bf16x8*)p;
      bf16x8 hi = *(const bf16x8*)(p + 16);
      bfr[j] = __builtin_shufflevector(lo, hi, 0,1,2,3,4,5,6,7,8,9,10,11,12,13,14,15);
    }
#pragma unroll
    for (int i = 0; i < 4; ++i)
#pragma unroll
      for (int j = 0; j < 2; ++j)
        acc[i][j] = __builtin_amdgcn_wmma_f32_16x16x32_bf16(
            false, af[i], false, bfr[j], (short)0, acc[i][j], false, false);

    if (kt + 1 < nk) {
      wait_async0();      // my wave's copies to buf^1 have landed in LDS
      __syncthreads();    // everyone's have; also all reads of buf are done
      buf ^= 1;
    }
  }

  // Epilogue: C layout lanes 0-15 -> M=r, lanes 16-31 -> M=r+8 (ISA 7.12.2)
  const int erow = (lane >> 4) << 3;
  const int ecol = lane & 15;
#pragma unroll
  for (int i = 0; i < 4; ++i)
#pragma unroll
    for (int j = 0; j < 2; ++j)
#pragma unroll
      for (int r = 0; r < 8; ++r) {
        const int mm = m0 + wm * 64 + i * 16 + erow + r;
        const int nn = n0 + wn * 32 + j * 16 + ecol;
        float v = alpha * acc[i][j][r];
        if (rowScale) v *= rowScale[(size_t)mm * rsStride];
        const size_t idx = (size_t)mm * ldc + nn;
        if (accum) Cb[idx] += v; else Cb[idx] = v;
      }
}

// ---------------------------------------------------------------------------
// Elementwise / reduction kernels
// ---------------------------------------------------------------------------
__global__ void k_embed(const int* __restrict__ ids, const float* __restrict__ WE,
                        float* __restrict__ h, int D) {
  const int t = blockIdx.x;
  const float* src = WE + (size_t)ids[t] * D;
  float* dst = h + (size_t)t * D;
  for (int i = threadIdx.x; i < D; i += 256) dst[i] = src[i];
}

__global__ void k_conv(const float* __restrict__ s, bf16* __restrict__ d, size_t n) {
  size_t i = ((size_t)blockIdx.x * 256 + threadIdx.x) * 4;
  if (i + 3 < n) {
    d[i + 0] = (bf16)s[i + 0];
    d[i + 1] = (bf16)s[i + 1];
    d[i + 2] = (bf16)s[i + 2];
    d[i + 3] = (bf16)s[i + 3];
  }
}

// fp32 [K,N] -> bf16 [N,K] (transpose), K%32==0, N%32==0
__global__ __launch_bounds__(256) void k_conv_T(const float* __restrict__ src,
                                                bf16* __restrict__ dst, int K, int N) {
  __shared__ float tile[32][33];
  const int kb = blockIdx.y * 32, nb = blockIdx.x * 32;
  const int tx = threadIdx.x & 31, ty = threadIdx.x >> 5;
  for (int i = ty; i < 32; i += 8)
    tile[i][tx] = src[(size_t)(kb + i) * N + nb + tx];
  __syncthreads();
  for (int i = ty; i < 32; i += 8)
    dst[(size_t)(nb + i) * K + kb + tx] = (bf16)tile[tx][i];
}

__global__ __launch_bounds__(256) void k_rmsnorm(const float* __restrict__ x,
                                                 const float* __restrict__ w,
                                                 bf16* __restrict__ outp, int D) {
  const float* row = x + (size_t)blockIdx.x * D;
  __shared__ float red[8];
  float ss = 0.f;
  for (int i = threadIdx.x; i < D; i += 256) { float v = row[i]; ss += v * v; }
  for (int o = 16; o; o >>= 1) ss += __shfl_xor(ss, o, 32);
  if ((threadIdx.x & 31) == 0) red[threadIdx.x >> 5] = ss;
  __syncthreads();
  float tot = red[0]+red[1]+red[2]+red[3]+red[4]+red[5]+red[6]+red[7];
  const float inv = rsqrtf(tot / (float)D + 1e-6f);
  bf16* orow = outp + (size_t)blockIdx.x * D;
  for (int i = threadIdx.x; i < D; i += 256) orow[i] = (bf16)(row[i] * inv * w[i]);
}

// RoPE on q/k + relayout to [B,H,S,HD] bf16; v -> transposed [B,H,HD,S] bf16
__global__ void k_qkv_post(const float* __restrict__ q, const float* __restrict__ k,
                           const float* __restrict__ v, bf16* __restrict__ qb,
                           bf16* __restrict__ kb, bf16* __restrict__ vtb,
                           int S, int Hh, int HDd) {
  const int s = blockIdx.x, hh = blockIdx.y, b = blockIdx.z, d = threadIdx.x;
  const int Dm = Hh * HDd;
  const size_t src  = (((size_t)b * S + s) * Dm) + hh * HDd + d;
  const size_t srcp = src + ((d < 32) ? 32 : -32);
  const int dr = d & 31;
  const float invf = __powf(10000.f, -((float)(2 * dr)) / (float)HDd);
  const float ang = (float)s * invf;
  const float c = __cosf(ang), sn = __sinf(ang);
  const float qx = q[src], qp = q[srcp];
  const float kx = k[src], kp = k[srcp];
  const float qv = (d < 32) ? (qx * c - qp * sn) : (qp * sn + qx * c);
  const float kv = (d < 32) ? (kx * c - kp * sn) : (kp * sn + kx * c);
  const size_t dq = ((((size_t)b * Hh) + hh) * S + s) * HDd + d;
  qb[dq] = (bf16)qv;
  kb[dq] = (bf16)kv;
  const size_t dv = ((((size_t)b * Hh) + hh) * HDd + d) * S + s;
  vtb[dv] = (bf16)v[src];
}

__global__ __launch_bounds__(256) void k_softmax_causal(const float* __restrict__ sc,
                                                        bf16* __restrict__ pr, int S) {
  const int qi = blockIdx.x;
  const size_t base = (size_t)blockIdx.y * S * S + (size_t)qi * S;
  const float* r = sc + base;
  bf16* o = pr + base;
  __shared__ float red[8];
  const int tid = threadIdx.x;
  float mx = -1e30f;
  for (int j = tid; j <= qi; j += 256) mx = fmaxf(mx, r[j]);
  for (int s = 16; s; s >>= 1) mx = fmaxf(mx, __shfl_xor(mx, s, 32));
  if ((tid & 31) == 0) red[tid >> 5] = mx;
  __syncthreads();
  const float m = fmaxf(fmaxf(fmaxf(red[0], red[1]), fmaxf(red[2], red[3])),
                        fmaxf(fmaxf(red[4], red[5]), fmaxf(red[6], red[7])));
  __syncthreads();
  float sm = 0.f;
  for (int j = tid; j <= qi; j += 256) sm += __expf(r[j] - m);
  for (int s = 16; s; s >>= 1) sm += __shfl_xor(sm, s, 32);
  if ((tid & 31) == 0) red[tid >> 5] = sm;
  __syncthreads();
  const float tot = red[0]+red[1]+red[2]+red[3]+red[4]+red[5]+red[6]+red[7];
  const float inv = 1.f / tot;
  for (int j = tid; j < S; j += 256)
    o[j] = (bf16)((j <= qi) ? __expf(r[j] - m) * inv : 0.f);
}

// [B,H,S,HD] fp32 -> [B,S,H*HD] bf16
__global__ void k_permute_attn(const float* __restrict__ in, bf16* __restrict__ outp,
                               int S, int Hh, int HDd) {
  const int s = blockIdx.x, hh = blockIdx.y, b = blockIdx.z, d = threadIdx.x;
  const size_t src = ((((size_t)b * Hh + hh) * S) + s) * HDd + d;
  const size_t dst = (((size_t)b * S + s) * (Hh * HDd)) + hh * HDd + d;
  outp[dst] = (bf16)in[src];
}

// Router: logits = xn @ Wr, top-2 softmax -> dense weight matrix wfull [T,E]
__global__ __launch_bounds__(256) void k_router(const bf16* __restrict__ xn,
                                                const float* __restrict__ Wr,
                                                float* __restrict__ wfull, int D) {
  const int wave = threadIdx.x >> 5, lane = threadIdx.x & 31;
  const int t = blockIdx.x * 8 + wave;
  const bf16* xr = xn + (size_t)t * D;
  float acc[CE] = {};
  for (int kk = lane; kk < D; kk += 32) {
    const float xv = (float)xr[kk];
    const float* wrow = Wr + (size_t)kk * CE;
#pragma unroll
    for (int e = 0; e < CE; ++e) acc[e] += xv * wrow[e];
  }
#pragma unroll
  for (int e = 0; e < CE; ++e)
    for (int o = 16; o; o >>= 1) acc[e] += __shfl_xor(acc[e], o, 32);
  if (lane == 0) {
    int i0 = 0; float v0 = acc[0];
    for (int e = 1; e < CE; ++e) if (acc[e] > v0) { v0 = acc[e]; i0 = e; }
    int i1 = -1; float v1 = -1e30f;
    for (int e = 0; e < CE; ++e) if (e != i0 && acc[e] > v1) { v1 = acc[e]; i1 = e; }
    const float w1 = __expf(v1 - v0);
    const float inv = 1.f / (1.f + w1);
    float* wo = wfull + (size_t)t * CE;
#pragma unroll
    for (int e = 0; e < CE; ++e) wo[e] = 0.f;
    wo[i0] = inv;
    wo[i1] = w1 * inv;
  }
}

__global__ void k_silu_mul(const float* __restrict__ g, const float* __restrict__ u,
                           bf16* __restrict__ act, size_t n) {
  size_t i = ((size_t)blockIdx.x * 256 + threadIdx.x) * 4;
  if (i + 3 < n) {
#pragma unroll
    for (int j = 0; j < 4; ++j) {
      const float gv = g[i + j];
      act[i + j] = (bf16)(gv / (1.f + __expf(-gv)) * u[i + j]);
    }
  }
}

// ---------------------------------------------------------------------------
// Host orchestration
// ---------------------------------------------------------------------------
extern "C" void kernel_launch(void* const* d_in, const int* in_sizes, int n_in,
                              void* d_out, int out_size, void* d_ws, size_t ws_size,
                              hipStream_t stream) {
  const int*   ids  = (const int*)d_in[0];
  const float* W_E  = (const float*)d_in[1];
  const float* ln1  = (const float*)d_in[2];
  const float* Wq   = (const float*)d_in[3];
  const float* Wk   = (const float*)d_in[4];
  const float* Wv   = (const float*)d_in[5];
  const float* Wo   = (const float*)d_in[6];
  const float* ln2  = (const float*)d_in[7];
  const float* Wr   = (const float*)d_in[8];
  const float* Wg   = (const float*)d_in[9];
  const float* Wu   = (const float*)d_in[10];
  const float* Wd   = (const float*)d_in[11];
  const float* finw = (const float*)d_in[12];
  float* out = (float*)d_out;

  // Workspace bump allocator
  size_t off = 0;
  auto alloc = [&](size_t bytes) {
    void* p = (char*)d_ws + off;
    off += (bytes + 255) & ~(size_t)255;
    return p;
  };
  bf16*  WEb   = (bf16*)alloc((size_t)CV * CD * 2);
  bf16*  WqT   = (bf16*)alloc((size_t)CD * CD * 2);
  bf16*  WkT   = (bf16*)alloc((size_t)CD * CD * 2);
  bf16*  WvT   = (bf16*)alloc((size_t)CD * CD * 2);
  bf16*  WoT   = (bf16*)alloc((size_t)CD * CD * 2);
  bf16*  WgTe  = (bf16*)alloc((size_t)CF * CD * 2);
  bf16*  WuTe  = (bf16*)alloc((size_t)CF * CD * 2);
  bf16*  WdTe  = (bf16*)alloc((size_t)CD * CF * 2);
  float* h     = (float*)alloc((size_t)CT * CD * 4);
  bf16*  xn    = (bf16*)alloc((size_t)CT * CD * 2);
  float* qf    = (float*)alloc((size_t)CT * CD * 4);
  float* kf    = (float*)alloc((size_t)CT * CD * 4);
  float* vf    = (float*)alloc((size_t)CT * CD * 4);
  bf16*  qb    = (bf16*)alloc((size_t)CT * CD * 2);
  bf16*  kb    = (bf16*)alloc((size_t)CT * CD * 2);
  bf16*  vtb   = (bf16*)alloc((size_t)CT * CD * 2);
  float* scf   = (float*)alloc((size_t)CB * CH * CS * CS * 4);
  bf16*  probs = (bf16*)alloc((size_t)CB * CH * CS * CS * 2);
  float* aobh  = (float*)alloc((size_t)CT * CD * 4);
  bf16*  attnb = (bf16*)alloc((size_t)CT * CD * 2);
  float* wfull = (float*)alloc((size_t)CT * CE * 4);
  float* gf    = (float*)alloc((size_t)CT * CF * 4);
  float* uf    = (float*)alloc((size_t)CT * CF * 4);
  bf16*  actb  = (bf16*)alloc((size_t)CT * CF * 2);
  bf16*  hn    = (bf16*)alloc((size_t)CT * CD * 2);

  auto gemm = [&](const bf16* A, long long sA, int lda, const bf16* Bt, long long sB,
                  int ldb, float* C, long long sC, int ldc, int M, int N, int Kd,
                  float alpha, const float* rs, int rsStride, int accum, int batch) {
    dim3 g((unsigned)(N / 64), (unsigned)(M / 256), (unsigned)batch);
    k_gemm_nt<<<g, dim3(256), 0, stream>>>(A, sA, lda, Bt, sB, ldb, C, sC, ldc,
                                           M, N, Kd, alpha, rs, rsStride, accum);
  };

  // Embedding + LM-head weight conversion (W_E is already [V,D] = BT layout)
  k_embed<<<dim3(CT), dim3(256), 0, stream>>>(ids, W_E, h, CD);
  k_conv<<<dim3((unsigned)((size_t)CV * CD / 1024)), dim3(256), 0, stream>>>(
      W_E, WEb, (size_t)CV * CD);

  for (int l = 0; l < CL; ++l) {
    // --- attention ---
    k_rmsnorm<<<dim3(CT), dim3(256), 0, stream>>>(h, ln1 + (size_t)l * CD, xn, CD);
    k_conv_T<<<dim3(CD/32, CD/32), dim3(256), 0, stream>>>(Wq + (size_t)l*CD*CD, WqT, CD, CD);
    k_conv_T<<<dim3(CD/32, CD/32), dim3(256), 0, stream>>>(Wk + (size_t)l*CD*CD, WkT, CD, CD);
    k_conv_T<<<dim3(CD/32, CD/32), dim3(256), 0, stream>>>(Wv + (size_t)l*CD*CD, WvT, CD, CD);
    k_conv_T<<<dim3(CD/32, CD/32), dim3(256), 0, stream>>>(Wo + (size_t)l*CD*CD, WoT, CD, CD);

    gemm(xn, 0, CD, WqT, 0, CD, qf, 0, CD, CT, CD, CD, 1.f, nullptr, 0, 0, 1);
    gemm(xn, 0, CD, WkT, 0, CD, kf, 0, CD, CT, CD, CD, 1.f, nullptr, 0, 0, 1);
    gemm(xn, 0, CD, WvT, 0, CD, vf, 0, CD, CT, CD, CD, 1.f, nullptr, 0, 0, 1);

    k_qkv_post<<<dim3(CS, CH, CB), dim3(CHD), 0, stream>>>(qf, kf, vf, qb, kb, vtb,
                                                           CS, CH, CHD);
    // scores = (1/sqrt(HD)) * q k^T   (batched over B*H)
    gemm(qb, (long long)CS * CHD, CHD, kb, (long long)CS * CHD, CHD,
         scf, (long long)CS * CS, CS, CS, CS, CHD, 0.125f, nullptr, 0, 0, CB * CH);
    k_softmax_causal<<<dim3(CS, CB * CH), dim3(256), 0, stream>>>(scf, probs, CS);
    // out = probs @ v   (BT = v^T [HD,S])
    gemm(probs, (long long)CS * CS, CS, vtb, (long long)CHD * CS, CS,
         aobh, (long long)CS * CHD, CHD, CS, CHD, CS, 1.f, nullptr, 0, 0, CB * CH);
    k_permute_attn<<<dim3(CS, CH, CB), dim3(CHD), 0, stream>>>(aobh, attnb, CS, CH, CHD);
    // h += attn_out @ Wo  (residual folded into epilogue)
    gemm(attnb, 0, CD, WoT, 0, CD, h, 0, CD, CT, CD, CD, 1.f, nullptr, 0, 1, 1);

    // --- MoE ---
    k_rmsnorm<<<dim3(CT), dim3(256), 0, stream>>>(h, ln2 + (size_t)l * CD, xn, CD);
    k_router<<<dim3(CT / 8), dim3(256), 0, stream>>>(xn, Wr + (size_t)l * CD * CE,
                                                     wfull, CD);
    for (int e = 0; e < CE; ++e) {
      const float* Wge = Wg + ((size_t)l * CE + e) * CD * CF;
      const float* Wue = Wu + ((size_t)l * CE + e) * CD * CF;
      const float* Wde = Wd + ((size_t)l * CE + e) * CF * CD;
      k_conv_T<<<dim3(CF/32, CD/32), dim3(256), 0, stream>>>(Wge, WgTe, CD, CF);
      k_conv_T<<<dim3(CF/32, CD/32), dim3(256), 0, stream>>>(Wue, WuTe, CD, CF);
      k_conv_T<<<dim3(CD/32, CF/32), dim3(256), 0, stream>>>(Wde, WdTe, CF, CD);
      gemm(xn, 0, CD, WgTe, 0, CD, gf, 0, CF, CT, CF, CD, 1.f, nullptr, 0, 0, 1);
      gemm(xn, 0, CD, WuTe, 0, CD, uf, 0, CF, CT, CF, CD, 1.f, nullptr, 0, 0, 1);
      k_silu_mul<<<dim3((unsigned)((size_t)CT * CF / 1024)), dim3(256), 0, stream>>>(
          gf, uf, actb, (size_t)CT * CF);
      // h += routing_weight[:,e] * (act @ Wd_e)
      gemm(actb, 0, CF, WdTe, 0, CF, h, 0, CD, CT, CD, CF, 1.f, wfull + e, CE, 1, 1);
    }
  }

  // Final norm + tied LM head
  k_rmsnorm<<<dim3(CT), dim3(256), 0, stream>>>(h, finw, hn, CD);
  gemm(hn, 0, CD, WEb, 0, CD, out, 0, CV, CT, CV, CD, 1.f, nullptr, 0, 0, 1);
}
